// DistillationSingleClassDetectionLoss_57827439673622
// MI455X (gfx1250) — compile-verified, matrix operand order checked by reading, not run on hardware
//
#include <hip/hip_runtime.h>
#include <hip/hip_bf16.h>
#include <math.h>

typedef __attribute__((ext_vector_type(16))) _Float16 v16h;
typedef __attribute__((ext_vector_type(8)))  float    v8f;

#define TPB 256
#define NWAVES (TPB / 32)
#define CENTER_RADIUS 2.5f
#define TOPK 10
#define IOU_W 3.0f
#define INF_COST 100000.0f
#define INVALID_COST 100000000.0f
#define EPSF 1e-7f

// ---------- helpers ----------
__device__ __forceinline__ unsigned f2ord(float x) {
    unsigned u = __float_as_uint(x);
    return (u & 0x80000000u) ? ~u : (u | 0x80000000u);
}
__device__ __forceinline__ float ord2f(unsigned k) {
    unsigned u = (k & 0x80000000u) ? (k & 0x7fffffffu) : ~k;
    return __uint_as_float(u);
}
__device__ __forceinline__ float iou_one(const float* pb, const float* gb) {
    float tlx = fmaxf(pb[0], gb[0]);
    float tly = fmaxf(pb[1], gb[1]);
    float brx = fminf(pb[2], gb[2]);
    float bry = fminf(pb[3], gb[3]);
    float w = fmaxf(brx - tlx, 0.f), h = fmaxf(bry - tly, 0.f);
    float inter = w * h;
    float aa = (pb[2] - pb[0]) * (pb[3] - pb[1]);
    float ab = (gb[2] - gb[0]) * (gb[3] - gb[1]);
    return inter / (aa + ab - inter + 1e-6f);
}
__device__ __forceinline__ float focal_term(float l, float t) {
    float p  = 1.f / (1.f + expf(-l));
    float ce = fmaxf(l, 0.f) - l * t + log1pf(expf(-fabsf(l)));
    float pt = p * t + (1.f - p) * (1.f - t);
    float at = 0.25f * t + 0.75f * (1.f - t);
    float om = 1.f - pt;
    return at * om * om * ce;
}
__device__ __forceinline__ float eiou_term(const float* p, const float* t) {
    float iw = fmaxf(fminf(p[2], t[2]) - fmaxf(p[0], t[0]), 0.f);
    float ih = fmaxf(fminf(p[3], t[3]) - fmaxf(p[1], t[1]), 0.f);
    float inter = iw * ih;
    float un  = (p[2] - p[0]) * (p[3] - p[1]) + (t[2] - t[0]) * (t[3] - t[1]) - inter;
    float iou = inter / (un + EPSF);
    float cw  = fmaxf(p[2], t[2]) - fminf(p[0], t[0]);
    float ch  = fmaxf(p[3], t[3]) - fminf(p[1], t[1]);
    float c2  = cw * cw + ch * ch + EPSF;
    float dx  = p[0] + p[2] - t[0] - t[2];
    float dy  = p[1] + p[3] - t[1] - t[3];
    float rho2 = (dx * dx + dy * dy) * 0.25f;
    float dw = (p[2] - p[0]) - (t[2] - t[0]);
    float dh = (p[3] - p[1]) - (t[3] - t[1]);
    return 1.f - iou + rho2 / c2 + (dw * dw) / (cw * cw + EPSF) + (dh * dh) / (ch * ch + EPSF);
}

// ---------- pass 1: per-image geometry (valid flag + in_both bitmask) ----------
__global__ void k_pass1_valid(const float* __restrict__ priors,
                              const float* __restrict__ tgts,
                              unsigned char* __restrict__ valid,
                              unsigned long long* __restrict__ inboth,
                              int A, int M) {
    int n = blockIdx.x;
    __shared__ float sg[64 * 4];
    for (int i = threadIdx.x; i < M * 4; i += blockDim.x)
        sg[i] = tgts[(size_t)n * M * 4 + i];
    __syncthreads();
    for (int a = threadIdx.x; a < A; a += blockDim.x) {
        const float* pr = priors + (size_t)a * 4;
        float sw = pr[2], sh = pr[3];
        float cx = pr[0] + sw * 0.5f, cy = pr[1] + sh * 0.5f;
        unsigned long long both = 0ull;
        bool anyv = false;
        for (int m = 0; m < M; ++m) {
            const float* g = sg + m * 4;
            bool ing = (cx > g[0]) && (cy > g[1]) && (cx < g[2]) && (cy < g[3]);
            float gcx = (g[0] + g[2]) * 0.5f, gcy = (g[1] + g[3]) * 0.5f;
            bool inc = (cx > gcx - CENTER_RADIUS * sw) && (cy > gcy - CENTER_RADIUS * sh) &&
                       (cx < gcx + CENTER_RADIUS * sw) && (cy < gcy + CENTER_RADIUS * sh);
            anyv |= (ing | inc);
            if (ing && inc) both |= (1ull << m);
        }
        valid[(size_t)n * A + a]  = anyv ? 1 : 0;
        inboth[(size_t)n * A + a] = both;
    }
}

// ---------- pass 2: per (gt, image, type) dynamic-k selection, column in LDS ----------
__global__ void __launch_bounds__(TPB, 1)
k_pass2_select(const float* __restrict__ s_dec,
               const float* __restrict__ t_dec,
               const float* __restrict__ tgts,
               const unsigned char* __restrict__ valid,
               const unsigned long long* __restrict__ inboth,
               unsigned long long* __restrict__ mask_s,
               unsigned long long* __restrict__ mask_t,
               int A, int M) {
    extern __shared__ unsigned char smem_dyn[];
    unsigned long long* keyIou  = (unsigned long long*)smem_dyn;
    unsigned long long* keyCost = (unsigned long long*)(smem_dyn + (size_t)A * 8);
    __shared__ unsigned long long sred[TPB];

    int m = blockIdx.x, n = blockIdx.y, ty = blockIdx.z;
    const float* dec = (ty == 0) ? s_dec : t_dec;
    unsigned long long* mask = (ty == 0) ? mask_s : mask_t;
    const float* g = tgts + ((size_t)n * M + m) * 4;
    float gb[4] = { g[0], g[1], g[2], g[3] };

    for (int a = threadIdx.x; a < A; a += blockDim.x) {
        const float* row = dec + ((size_t)n * A + a) * 5;
        if (a + (int)blockDim.x < A)
            __builtin_prefetch(dec + ((size_t)n * A + a + blockDim.x) * 5, 0, 3);
        float score = row[0];
        float pb[4] = { row[1], row[2], row[3], row[4] };
        float vi = valid[(size_t)n * A + a] ? 1.f : 0.f;
        float iou = iou_one(pb, gb) * vi;
        float cls = -0.5f * logf(fmaxf(score, 1e-12f));
        float cost = cls + IOU_W * (-logf(iou + EPSF));
        if (!((inboth[(size_t)n * A + a] >> m) & 1ull)) cost += INF_COST;
        if (vi == 0.f) cost += INVALID_COST;
        keyIou[a]  = ((unsigned long long)f2ord(iou)  << 32) | (unsigned)a;
        keyCost[a] = ((unsigned long long)f2ord(cost) << 32) | (unsigned)a;
    }
    __syncthreads();

    // top-10 IoU sum -> dynamic k
    unsigned long long prev = 0xFFFFFFFFFFFFFFFFull;
    float ksum = 0.f;
    for (int r = 0; r < TOPK; ++r) {
        unsigned long long best = 0ull;
        for (int a = threadIdx.x; a < A; a += blockDim.x) {
            unsigned long long k = keyIou[a];
            if (k < prev && k > best) best = k;
        }
        sred[threadIdx.x] = best;
        __syncthreads();
        for (int s = TPB / 2; s > 0; s >>= 1) {
            if (threadIdx.x < s) {
                unsigned long long o = sred[threadIdx.x + s];
                if (o > sred[threadIdx.x]) sred[threadIdx.x] = o;
            }
            __syncthreads();
        }
        best = sred[0];
        __syncthreads();
        if (best == 0ull) break;  // uniform
        ksum += ord2f((unsigned)(best >> 32));
        prev = best;
    }
    int dk = (int)ksum;
    if (dk < 1) dk = 1;
    if (dk > TOPK) dk = TOPK;

    // select dk smallest cost keys (stable: key = cost|index)
    prev = 0ull;
    for (int r = 0; r < dk; ++r) {
        unsigned long long best = 0xFFFFFFFFFFFFFFFFull;
        for (int a = threadIdx.x; a < A; a += blockDim.x) {
            unsigned long long k = keyCost[a];
            if (k > prev && k < best) best = k;
        }
        sred[threadIdx.x] = best;
        __syncthreads();
        for (int s = TPB / 2; s > 0; s >>= 1) {
            if (threadIdx.x < s) {
                unsigned long long o = sred[threadIdx.x + s];
                if (o < sred[threadIdx.x]) sred[threadIdx.x] = o;
            }
            __syncthreads();
        }
        best = sred[0];
        __syncthreads();
        if (best == 0xFFFFFFFFFFFFFFFFull) break;  // uniform
        if (threadIdx.x == 0) {
            unsigned a = (unsigned)(best & 0xffffffffull);
            atomicOr(&mask[(size_t)n * A + a], 1ull << m);
        }
        prev = best;
    }
}

// ---------- pass 3: resolve multi-matches -> gt_inds / matched_iou ----------
__global__ void k_pass3_resolve(const float* __restrict__ s_dec,
                                const float* __restrict__ t_dec,
                                const float* __restrict__ tgts,
                                const unsigned char* __restrict__ valid,
                                const unsigned long long* __restrict__ inboth,
                                const unsigned long long* __restrict__ mask_s,
                                const unsigned long long* __restrict__ mask_t,
                                int* __restrict__ gt_s, int* __restrict__ gt_t,
                                float* __restrict__ miou_s,
                                int A, int M) {
    int n = blockIdx.x, ty = blockIdx.y;
    const float* dec = ty ? t_dec : s_dec;
    const unsigned long long* mask = ty ? mask_t : mask_s;
    int* gout = ty ? gt_t : gt_s;
    __shared__ float sgt[64 * 4];
    for (int i = threadIdx.x; i < M * 4; i += blockDim.x)
        sgt[i] = tgts[(size_t)n * M * 4 + i];
    __syncthreads();
    for (int a = threadIdx.x; a < A; a += blockDim.x) {
        unsigned long long mk = mask[(size_t)n * A + a];
        int cnt = __popcll(mk);
        int gi = 0;
        float mi = 0.f;
        if (cnt > 0) {
            const float* row = dec + ((size_t)n * A + a) * 5;
            float pb[4] = { row[1], row[2], row[3], row[4] };
            float vi = valid[(size_t)n * A + a] ? 1.f : 0.f;
            if (cnt == 1) {
                int m0 = __ffsll((unsigned long long)mk) - 1;
                gi = m0 + 1;
                mi = iou_one(pb, sgt + m0 * 4) * vi;
            } else {
                float cls = -0.5f * logf(fmaxf(row[0], 1e-12f));
                unsigned long long both = inboth[(size_t)n * A + a];
                float bestc = 3.4e38f, besti = 0.f;
                int bestm = 0;
                for (int m = 0; m < M; ++m) {
                    float iou = iou_one(pb, sgt + m * 4) * vi;
                    float c = cls + IOU_W * (-logf(iou + EPSF));
                    if (!((both >> m) & 1ull)) c += INF_COST;
                    if (vi == 0.f) c += INVALID_COST;
                    if (c < bestc) { bestc = c; bestm = m; besti = iou; }
                }
                gi = bestm + 1;
                mi = besti;
            }
        }
        gout[(size_t)n * A + a] = gi;
        if (ty == 0) miou_s[(size_t)n * A + a] = mi;
    }
}

// ---------- pass 4: focal + EIoU sums reduced via v_wmma_f32_16x16x32_f16 ----------
__global__ void __launch_bounds__(TPB, 1)
k_pass4_loss(const float* __restrict__ s_pred,
             const float* __restrict__ t_pred,
             const float* __restrict__ s_dec,
             const float* __restrict__ t_dec,
             const float* __restrict__ tgts,
             const int* __restrict__ gt_s,
             const int* __restrict__ gt_t,
             const float* __restrict__ miou_s,
             float* __restrict__ per_image,
             int A, int M) {
    int n = blockIdx.x;
    int lane = threadIdx.x & 31;
    int wid  = threadIdx.x >> 5;

    v8f acc0 = {}, acc1 = {}, acc2 = {}, acc3 = {}, acc4 = {}, acc5 = {};
    v16h ones;
    for (int i = 0; i < 16; ++i) ones[i] = (_Float16)1.0f;

    int chunks = (A + 511) / 512;
    for (int c = wid; c < chunks; c += NWAVES) {   // uniform per wave -> EXEC all-1 at WMMA
        int base = c * 512 + lane * 16;
        v16h q0, q1, q2, q3, q4, q5;
        for (int i = 0; i < 16; ++i) {
            float v0 = 0.f, v1 = 0.f, v2 = 0.f, v3 = 0.f, v4 = 0.f, v5 = 0.f;
            int a = base + i;
            if (a < A) {
                size_t na = (size_t)n * A + a;
                float l   = s_pred[na * 5];
                int   g   = gt_s[na];
                float ct  = (g > 0) ? miou_s[na] : 0.f;
                v0 = focal_term(l, ct);
                float tl = t_pred[na * 5];
                v1 = focal_term(l, 1.f / (1.f + expf(-tl)));
                const float* sr = s_dec + na * 5;
                float pb[4] = { sr[1], sr[2], sr[3], sr[4] };
                int ti = (g > 0) ? (g - 1) : 0;
                const float* gbp = tgts + ((size_t)n * M + ti) * 4;
                float gbb[4] = { gbp[0], gbp[1], gbp[2], gbp[3] };
                float sp = (g > 0) ? 1.f : 0.f;
                v2 = eiou_term(pb, gbb) * sp;
                v3 = sp;
                const float* tr = t_dec + na * 5;
                float tbb[4] = { tr[1], tr[2], tr[3], tr[4] };
                float tp = (gt_t[na] > 0) ? 1.f : 0.f;
                v4 = eiou_term(pb, tbb) * tp;
                v5 = tp;
            }
            q0[i] = (_Float16)v0; q1[i] = (_Float16)v1; q2[i] = (_Float16)v2;
            q3[i] = (_Float16)v3; q4[i] = (_Float16)v4; q5[i] = (_Float16)v5;
        }
        // D[i][j] = rowsum_i ; total over C = 16 * sum(tile). f32 accumulation.
        acc0 = __builtin_amdgcn_wmma_f32_16x16x32_f16(false, q0, false, ones, (short)0, acc0, false, false);
        acc1 = __builtin_amdgcn_wmma_f32_16x16x32_f16(false, q1, false, ones, (short)0, acc1, false, false);
        acc2 = __builtin_amdgcn_wmma_f32_16x16x32_f16(false, q2, false, ones, (short)0, acc2, false, false);
        acc3 = __builtin_amdgcn_wmma_f32_16x16x32_f16(false, q3, false, ones, (short)0, acc3, false, false);
        acc4 = __builtin_amdgcn_wmma_f32_16x16x32_f16(false, q4, false, ones, (short)0, acc4, false, false);
        acc5 = __builtin_amdgcn_wmma_f32_16x16x32_f16(false, q5, false, ones, (short)0, acc5, false, false);
    }

    float t0 = 0.f, t1 = 0.f, t2 = 0.f, t3 = 0.f, t4 = 0.f, t5 = 0.f;
    for (int r = 0; r < 8; ++r) {
        t0 += acc0[r]; t1 += acc1[r]; t2 += acc2[r];
        t3 += acc3[r]; t4 += acc4[r]; t5 += acc5[r];
    }
    for (int off = 16; off > 0; off >>= 1) {
        t0 += __shfl_xor(t0, off, 32);
        t1 += __shfl_xor(t1, off, 32);
        t2 += __shfl_xor(t2, off, 32);
        t3 += __shfl_xor(t3, off, 32);
        t4 += __shfl_xor(t4, off, 32);
        t5 += __shfl_xor(t5, off, 32);
    }
    __shared__ float wsum[NWAVES][6];
    if (lane == 0) {
        wsum[wid][0] = t0; wsum[wid][1] = t1; wsum[wid][2] = t2;
        wsum[wid][3] = t3; wsum[wid][4] = t4; wsum[wid][5] = t5;
    }
    __syncthreads();
    if (threadIdx.x == 0) {
        float q[6] = { 0, 0, 0, 0, 0, 0 };
        for (int w = 0; w < NWAVES; ++w)
            for (int j = 0; j < 6; ++j) q[j] += wsum[w][j];
        for (int j = 0; j < 6; ++j) q[j] *= (1.f / 16.f);  // undo 16x column replication
        float target_conf  = q[0] / (float)A;
        float teacher_conf = q[1] / (float)A;
        float target_bbox  = q[2] / fmaxf(q[3], 1.f);
        float teacher_bbox = q[4] / fmaxf(q[5], 1.f);
        float conf = 0.25f * target_conf + 0.75f * teacher_conf;
        float bbox = 0.25f * target_bbox + 0.75f * teacher_bbox;
        per_image[n] = conf + 5.f * bbox;
    }
}

// ---------- pass 5: mean over images ----------
__global__ void k_pass5_final(const float* __restrict__ per_image,
                              float* __restrict__ out, int N) {
    if (threadIdx.x == 0) {
        float s = 0.f;
        for (int i = 0; i < N; ++i) s += per_image[i];
        out[0] = s / (float)N;
    }
}

extern "C" void kernel_launch(void* const* d_in, const int* in_sizes, int n_in,
                              void* d_out, int out_size, void* d_ws, size_t ws_size,
                              hipStream_t stream) {
    const float* s_pred = (const float*)d_in[0];
    const float* priors = (const float*)d_in[1];
    const float* s_dec  = (const float*)d_in[2];
    const float* t_pred = (const float*)d_in[3];
    const float* t_dec  = (const float*)d_in[5];
    const float* tgts   = (const float*)d_in[6];

    int A = in_sizes[1] / 4;
    int N = in_sizes[0] / (A * 5);
    int M = in_sizes[6] / (N * 4);

    size_t NA = (size_t)N * A;
    unsigned char* base = (unsigned char*)d_ws;
    size_t off = 0;
    unsigned long long* inboth = (unsigned long long*)(base + off); off += NA * 8;
    unsigned long long* mask_s = (unsigned long long*)(base + off); off += NA * 8;
    unsigned long long* mask_t = (unsigned long long*)(base + off); off += NA * 8;
    int*   gt_s      = (int*)(base + off);   off += NA * 4;
    int*   gt_t      = (int*)(base + off);   off += NA * 4;
    float* miou_s    = (float*)(base + off); off += NA * 4;
    float* per_image = (float*)(base + off); off += (size_t)N * 4;
    unsigned char* valid = base + off;       off += NA;

    // zero the match bitmasks (atomicOr targets) every launch
    (void)hipMemsetAsync(mask_s, 0, NA * 16, stream);

    k_pass1_valid<<<dim3(N), dim3(TPB), 0, stream>>>(priors, tgts, valid, inboth, A, M);

    size_t smem2 = (size_t)A * 16;  // keyIou + keyCost columns in LDS (CDNA5: 320KB/WGP)
    k_pass2_select<<<dim3(M, N, 2), dim3(TPB), smem2, stream>>>(
        s_dec, t_dec, tgts, valid, inboth, mask_s, mask_t, A, M);

    k_pass3_resolve<<<dim3(N, 2), dim3(TPB), 0, stream>>>(
        s_dec, t_dec, tgts, valid, inboth, mask_s, mask_t, gt_s, gt_t, miou_s, A, M);

    k_pass4_loss<<<dim3(N), dim3(TPB), 0, stream>>>(
        s_pred, t_pred, s_dec, t_dec, tgts, gt_s, gt_t, miou_s, per_image, A, M);

    k_pass5_final<<<dim3(1), dim3(32), 0, stream>>>(per_image, (float*)d_out, N);
}
